// MultiHeadSelfAttention_11871289606668
// MI455X (gfx1250) — matrix operand order repaired and checked by reading, not compiled, not run on hardware
//
#include <hip/hip_runtime.h>

// Problem constants (fixed by reference): B=4, N=2048, C=1024, H=16, D=64
#define SEQ     2048
#define NBATCH  4
#define NTOK    (NBATCH * SEQ)     // 8192
#define CDIM    1024
#define NHEAD   16
#define HDIM    64
// Q scale with log2(e) folded in: softmax runs in exp2 domain.
#define QSCALE  (0.125f * 1.4426950408889634f)
#define LN_EPS  1e-6f

typedef __attribute__((ext_vector_type(16))) _Float16 v16h;
typedef __attribute__((ext_vector_type(8)))  _Float16 v8h;
typedef __attribute__((ext_vector_type(2)))  _Float16 v2h;
typedef __attribute__((ext_vector_type(8)))  float    v8f;
typedef int gint4 __attribute__((vector_size(16)));   // matches async builtin

// ---------------------------------------------------------------------------
// CDNA5 async global->LDS copy (ASYNCcnt-tracked), with portable fallback.
// ---------------------------------------------------------------------------
#if defined(__has_builtin)
#if __has_builtin(__builtin_amdgcn_global_load_async_to_lds_b128) && \
    __has_builtin(__builtin_amdgcn_s_wait_asynccnt)
#define HAVE_ASYNC_LDS 1
#endif
#if __has_builtin(__builtin_amdgcn_tensor_load_to_lds) && \
    __has_builtin(__builtin_amdgcn_s_wait_tensorcnt)
#define HAVE_TDM 1
#endif
#endif
#ifndef HAVE_ASYNC_LDS
#define HAVE_ASYNC_LDS 0
#endif
#ifndef HAVE_TDM
#define HAVE_TDM 0
#endif

__device__ __forceinline__ void cp16(_Float16* ldst, const _Float16* gsrc) {
#if HAVE_ASYNC_LDS
  __builtin_amdgcn_global_load_async_to_lds_b128(
      (gint4 __attribute__((address_space(1)))*)gsrc,
      (gint4 __attribute__((address_space(3)))*)ldst, /*offset=*/0, /*cpol=*/0);
#else
  *(uint4*)ldst = *(const uint4*)gsrc;
#endif
}

#if HAVE_ASYNC_LDS
#define WAIT_ASYNC(n) __builtin_amdgcn_s_wait_asynccnt(n)
#else
#define WAIT_ASYNC(n)
#endif

// ---------------------------------------------------------------------------
// Tensor Data Mover: one-instruction DMA of a strided 2D f16 tile into LDS.
// D# per CDNA5 ISA §8 (group0: count/lds_addr/global_addr/type=2;
// group1: data_size=2B, tensor dims, tile dims, dim0 stride; groups 2-3: 0).
// This toolchain's builtin takes 6 args:
//   (uint32x4 g0, int32x8 g1, int32x4 g2, int32x4 g3, int32x8 g4, i32 cpol)
// ---------------------------------------------------------------------------
#if HAVE_TDM
typedef unsigned int tdmv4u __attribute__((ext_vector_type(4)));
typedef int tdmv8i __attribute__((ext_vector_type(8)));
typedef int tdmv4i __attribute__((ext_vector_type(4)));

__device__ __forceinline__ unsigned lds_off(void* p) {
  return (unsigned)(unsigned long long)
      (void __attribute__((address_space(3)))*)p;
}

__device__ __forceinline__ void tdm_load_2d(unsigned lds_addr,
                                            const void* gsrc,
                                            unsigned tile_d0, unsigned tile_d1,
                                            unsigned tens_d0, unsigned tens_d1,
                                            unsigned stride0) {
  unsigned long long ga = (unsigned long long)gsrc;
  tdmv4u g0 = {1u,                              // count=1 (user descriptor)
               lds_addr,                        // lds_addr [63:32]
               (unsigned)ga,                    // global_addr [95:64]
               (unsigned)((ga >> 32) & 0x01ffffffu) | 0x80000000u}; // type=2
  tdmv8i g1;
  g1[0] = 0x00010000;                           // wg_mask=0, data_size=2B
  g1[1] = (int)((tens_d0 & 0xffffu) << 16);     // tensor_dim0[15:0]
  g1[2] = (int)((tens_d0 >> 16) | ((tens_d1 & 0xffffu) << 16));
  g1[3] = (int)((tens_d1 >> 16) | ((tile_d0 & 0xffffu) << 16));
  g1[4] = (int)(tile_d1 & 0xffffu);             // tile_dim1 (tile_dim2=0)
  g1[5] = (int)stride0;                         // tensor_dim0_stride[31:0]
  g1[6] = 0;
  g1[7] = 0;
  tdmv4i z4 = {0, 0, 0, 0};
  tdmv8i z8 = {0, 0, 0, 0, 0, 0, 0, 0};
  __builtin_amdgcn_tensor_load_to_lds(g0, g1, z4, z4, z8, 0);
}
#endif

// ---------------------------------------------------------------------------
// WMMA fragment helpers (wave32, per CDNA5 ISA 7.12.2 layouts)
// ---------------------------------------------------------------------------
__device__ __forceinline__ v16h frag_a_lds(const _Float16* base, int stride) {
  int lane = threadIdx.x & 31;
  int row  = lane & 15;
  int kb   = (lane & 16) >> 1;          // 0 or 8
  const _Float16* p = base + row * stride + kb;
  v8h lo = *(const v8h*)(p);
  v8h hi = *(const v8h*)(p + 16);
  v16h a;
#pragma unroll
  for (int j = 0; j < 8; ++j) { a[j] = lo[j]; a[8 + j] = hi[j]; }
  return a;
}

__device__ __forceinline__ v16h frag_bt_lds(const _Float16* base, int stride) {
  int lane = threadIdx.x & 31;
  int n    = lane & 15;
  int kb   = (lane & 16) >> 1;          // 0 or 8
  const _Float16* p = base + n * stride + kb;
  v8h lo = *(const v8h*)(p);
  v8h hi = *(const v8h*)(p + 16);
  v16h b;
#pragma unroll
  for (int j = 0; j < 8; ++j) { b[j] = lo[j]; b[8 + j] = hi[j]; }
  return b;
}

__device__ __forceinline__ v8f wmma16x16x32(v16h a, v16h b, v8f c) {
  return __builtin_amdgcn_wmma_f32_16x16x32_f16(
      false, a, false, b, (short)0, c, false, false);
}

__device__ __forceinline__ unsigned pk_rtz(float a, float b) {
  return __builtin_bit_cast(unsigned, __builtin_amdgcn_cvt_pkrtz(a, b));
}

__device__ __forceinline__ unsigned pk_max(unsigned x, unsigned y) {
  v2h a = __builtin_bit_cast(v2h, x);
  v2h b = __builtin_bit_cast(v2h, y);
  return __builtin_bit_cast(unsigned, __builtin_elementwise_max(a, b));
}

// ---------------------------------------------------------------------------
// Prep kernel A: elementwise f32 -> f16 convert (for x)
// ---------------------------------------------------------------------------
__global__ __launch_bounds__(256) void cvt_f16_kernel(
    const float* __restrict__ src, _Float16* __restrict__ dst) {
  size_t i = ((size_t)blockIdx.x * 256 + threadIdx.x) * 4;
  float4 v = *(const float4*)(src + i);
  dst[i + 0] = (_Float16)v.x;
  dst[i + 1] = (_Float16)v.y;
  dst[i + 2] = (_Float16)v.z;
  dst[i + 3] = (_Float16)v.w;
}

// ---------------------------------------------------------------------------
// Prep kernel B: transpose+convert: src f32 [R][C] -> dst f16 [C][R]
// ---------------------------------------------------------------------------
__global__ __launch_bounds__(256) void transpose_cvt_kernel(
    const float* __restrict__ src, _Float16* __restrict__ dst, int R, int C) {
  __shared__ _Float16 t[32][33];
  int nbc = C >> 5;
  int r0 = (blockIdx.x / nbc) << 5;
  int c0 = (blockIdx.x % nbc) << 5;
  int lr = threadIdx.x >> 5;
  int lc = threadIdx.x & 31;
#pragma unroll
  for (int i = 0; i < 4; ++i)
    t[lr + i * 8][lc] = (_Float16)src[(size_t)(r0 + lr + i * 8) * C + c0 + lc];
  __syncthreads();
#pragma unroll
  for (int i = 0; i < 4; ++i)
    dst[(size_t)(c0 + lr + i * 8) * R + r0 + lc] = t[lc][lr + i * 8];
}

// ---------------------------------------------------------------------------
// GEMM: C[M,N] = A[M,K] @ Bt[N,K]^T + bias[N]   (all-f16 operands, f32 accum)
// 256 thr / 8 waves, block tile 128x256, wave tile 64x64 (acc[4][4]),
// K-step 64, double-buffered; tiles staged by the Tensor Data Mover
// (2 tensor_load_to_lds per K-step, issued by wave 0) with async fallback.
// ---------------------------------------------------------------------------
__global__ __launch_bounds__(256) void gemm_f16_bias_kernel(
    const _Float16* __restrict__ Ah, const _Float16* __restrict__ Bt,
    const float* __restrict__ bias, float* __restrict__ Cp,
    int M, int K, int N) {
  __shared__ _Float16 As[2][128 * 64];
  __shared__ _Float16 Bs[2][256 * 64];

  int tid = threadIdx.x, lane = tid & 31, wid = tid >> 5;
  int nb = N >> 8;                      // N / 256
  int m0 = (blockIdx.x / nb) << 7;
  int n0 = (blockIdx.x % nb) << 8;
  int waveM = wid >> 2, waveN = wid & 3;
  const _Float16* Ag = Ah + (size_t)m0 * K;
  const _Float16* Bg = Bt + (size_t)n0 * K;

  auto stage = [&](int buf, int kk) {
#if HAVE_TDM
    if (tid == 0) {
      tdm_load_2d(lds_off(&As[buf][0]), Ag + kk, 64, 128, (unsigned)K, 128,
                  (unsigned)K);
      tdm_load_2d(lds_off(&Bs[buf][0]), Bg + kk, 64, 256, (unsigned)K, 256,
                  (unsigned)K);
    }
#else
#pragma unroll
    for (int i = 0; i < 4; ++i) {       // A: 128x64 halves = 1024 x 16B
      int idx = i * 256 + tid;
      int r = idx >> 3, j = (idx & 7) << 3;
      cp16(&As[buf][r * 64 + j], Ag + (size_t)r * K + kk + j);
    }
#pragma unroll
    for (int i = 0; i < 8; ++i) {       // B: 256x64 halves = 2048 x 16B
      int idx = i * 256 + tid;
      int r = idx >> 3, j = (idx & 7) << 3;
      cp16(&Bs[buf][r * 64 + j], Bg + (size_t)r * K + kk + j);
    }
#endif
  };

  v8f acc[4][4] = {};
  int niter = K >> 6;
  stage(0, 0);
  for (int kc = 0; kc < niter; ++kc) {
    int cur = kc & 1;
    if (kc + 1 < niter) {
      stage(cur ^ 1, (kc + 1) << 6);
#if HAVE_TDM
      if (wid == 0) __builtin_amdgcn_s_wait_tensorcnt(2);
#else
      WAIT_ASYNC(12);
#endif
    } else {
#if HAVE_TDM
      if (wid == 0) __builtin_amdgcn_s_wait_tensorcnt(0);
#else
      WAIT_ASYNC(0);
#endif
    }
    __syncthreads();
    const _Float16* Aw = As[cur] + (waveM * 64) * 64;
    const _Float16* Bw = Bs[cur] + (waveN * 64) * 64;
#pragma unroll
    for (int dk = 0; dk < 64; dk += 32) {
      v16h af[4];
#pragma unroll
      for (int mt = 0; mt < 4; ++mt)
        af[mt] = frag_a_lds(Aw + (mt * 16) * 64 + dk, 64);
#pragma unroll
      for (int nt = 0; nt < 4; ++nt) {
        v16h bf = frag_bt_lds(Bw + (nt * 16) * 64 + dk, 64);
#pragma unroll
        for (int mt = 0; mt < 4; ++mt)
          acc[mt][nt] = wmma16x16x32(af[mt], bf, acc[mt][nt]);
      }
    }
    __syncthreads();
  }

  int halfb = (lane >> 4) << 3;
  int colo  = lane & 15;
#pragma unroll
  for (int mt = 0; mt < 4; ++mt) {
#pragma unroll
    for (int nt = 0; nt < 4; ++nt) {
      int col  = n0 + waveN * 64 + nt * 16 + colo;
      float bi = bias[col];
#pragma unroll
      for (int i = 0; i < 8; ++i) {
        int row = m0 + waveM * 64 + mt * 16 + halfb + i;
        Cp[(size_t)row * N + col] = acc[mt][nt][i] + bi;
      }
    }
  }
}

// ---------------------------------------------------------------------------
// LayerNorm(q), LayerNorm(k), q*(SCALE*log2e), head-split -> f16.
// Q,K -> [B,H,N,D]; V -> [B,H,D,N] (pre-transposed for the PV WMMA B-operand)
// ---------------------------------------------------------------------------
__device__ __forceinline__ void ln_part(const float* __restrict__ src,
                                        const float* __restrict__ gamma,
                                        const float* __restrict__ beta,
                                        float scl, _Float16* __restrict__ dst,
                                        int b, int n, int tid, float* red) {
  float x[4];
  float s = 0.f, s2 = 0.f;
#pragma unroll
  for (int i = 0; i < 4; ++i) {
    x[i] = src[tid + i * 256];
    s += x[i];
    s2 += x[i] * x[i];
  }
#pragma unroll
  for (int m = 1; m < 32; m <<= 1) {
    s  += __shfl_xor(s, m, 32);
    s2 += __shfl_xor(s2, m, 32);
  }
  int wid = tid >> 5, lane = tid & 31;
  __syncthreads();
  if (lane == 0) { red[wid] = s; red[8 + wid] = s2; }
  __syncthreads();
  float S = 0.f, S2 = 0.f;
#pragma unroll
  for (int w = 0; w < 8; ++w) { S += red[w]; S2 += red[8 + w]; }
  float mu  = S * (1.f / 1024.f);
  float var = S2 * (1.f / 1024.f) - mu * mu;
  float inv = rsqrtf(var + LN_EPS);
#pragma unroll
  for (int i = 0; i < 4; ++i) {
    int c   = tid + i * 256;
    float y = ((x[i] - mu) * inv * gamma[c] + beta[c]) * scl;
    int h = c >> 6, d = c & 63;
    dst[(((size_t)(b * NHEAD + h)) * SEQ + n) * HDIM + d] = (_Float16)y;
  }
}

__global__ __launch_bounds__(256) void qk_norm_kernel(
    const float* __restrict__ qkv,
    const float* __restrict__ qg, const float* __restrict__ qb,
    const float* __restrict__ kg, const float* __restrict__ kbt,
    _Float16* __restrict__ Qh, _Float16* __restrict__ Kh,
    _Float16* __restrict__ Vt) {
  __shared__ float red[16];
  int token = blockIdx.x;
  int b = token >> 11;
  int n = token & 2047;
  int tid = threadIdx.x;
  const float* row = qkv + (size_t)token * (3 * CDIM);

  ln_part(row,        qg, qb, QSCALE, Qh, b, n, tid, red);
  ln_part(row + CDIM, kg, kbt, 1.0f,  Kh, b, n, tid, red);
#pragma unroll
  for (int i = 0; i < 4; ++i) {
    int c = tid + i * 256;
    int h = c >> 6, d = c & 63;
    Vt[((size_t)(b * NHEAD + h) * HDIM + d) * SEQ + n] =
        (_Float16)row[2 * CDIM + c];
  }
}

// ---------------------------------------------------------------------------
// Flash attention: 128 thr (4 waves), 64 query rows / block, 64-key chunks,
// double-buffered staging: K via per-lane async loads (permuted rows so the
// 4 probabilities per (row,c) pack into one b64 P store), V via TDM (64x64
// tile from the 2048-stride transposed V tensor). Softmax in exp2 domain;
// row sums via P @ ones WMMA; row max via packed-f16 shuffles.
// 18 WMMA per wave per chunk.
// ---------------------------------------------------------------------------
#if HAVE_TDM
#define ATTN_ACNT 4
#else
#define ATTN_ACNT 8
#endif

__global__ __launch_bounds__(128) void attn_kernel(
    const _Float16* __restrict__ Qh, const _Float16* __restrict__ Kh,
    const _Float16* __restrict__ Vt, _Float16* __restrict__ Of) {
  __shared__ _Float16 Qs[64 * 64];      // [qrow][d]
  __shared__ _Float16 Ks[2][64 * 64];   // [perm(key)][d]
  __shared__ _Float16 Vs[2][64 * 64];   // [d][key]
  __shared__ _Float16 Ps[4][16 * 64];   // per-wave P scratch [row][key]

  int bh  = blockIdx.x >> 5;
  int qt  = blockIdx.x & 31;
  int tid = threadIdx.x;
  int wid = tid >> 5, lane = tid & 31;

  const _Float16* Qg = Qh + ((size_t)bh * SEQ + (size_t)qt * 64) * HDIM;
  const _Float16* Kg = Kh + (size_t)bh * SEQ * HDIM;
  const _Float16* Vg = Vt + (size_t)bh * HDIM * SEQ;

  {
    const uint4* src = (const uint4*)Qg;
    uint4* dst = (uint4*)Qs;
#pragma unroll
    for (int i = 0; i < 4; ++i) dst[i * 128 + tid] = src[i * 128 + tid];
  }

  auto stage = [&](int buf, int k0) {
#pragma unroll
    for (int i = 0; i < 4; ++i) {
      int idx = i * 128 + tid;          // 0..511
      int s = idx >> 3, j = (idx & 7) << 3;
      // K: slot s holds key 4*(s&15) + (s>>4)  (row-granular permutation)
      int key = ((s & 15) << 2) | (s >> 4);
      cp16(&Ks[buf][s * 64 + j], Kg + (size_t)(k0 + key) * HDIM + j);
#if !HAVE_TDM
      cp16(&Vs[buf][s * 64 + j], Vg + (size_t)s * SEQ + k0 + j);
#endif
    }
#if HAVE_TDM
    if (tid == 0)
      tdm_load_2d(lds_off(&Vs[buf][0]), Vg + k0, 64, 64, SEQ, 64, SEQ);
#endif
  };

  float mS[8], lS[8], alv[8];
  v8f acc[4] = {};
#pragma unroll
  for (int i = 0; i < 8; ++i) { mS[i] = -3.0e38f; lS[i] = 0.f; }

  v16h ones;
#pragma unroll
  for (int j = 0; j < 16; ++j) ones[j] = (_Float16)1.0f;

  int halfb = (lane >> 4) << 3;
  int colo  = lane & 15;
  const _Float16* Qw = Qs + (wid * 16) * 64;

  int niter = SEQ / 64;                 // 32
  stage(0, 0);
  for (int kc = 0; kc < niter; ++kc) {
    int cur = kc & 1;
    if (kc + 1 < niter) {
      stage(cur ^ 1, (kc + 1) * 64);
      WAIT_ASYNC(ATTN_ACNT);
#if HAVE_TDM
      if (wid == 0) __builtin_amdgcn_s_wait_tensorcnt(1);
#endif
    } else {
      WAIT_ASYNC(0);
#if HAVE_TDM
      if (wid == 0) __builtin_amdgcn_s_wait_tensorcnt(0);
#endif
    }
    __syncthreads();

    // S = Q (16x64) @ K^T -> four 16x16 col-tiles (permuted key order)
    v8f s[4] = {};
#pragma unroll
    for (int dk = 0; dk < HDIM; dk += 32) {
      v16h aq = frag_a_lds(Qw + dk, 64);
#pragma unroll
      for (int nt = 0; nt < 4; ++nt) {
        v16h bk = frag_bt_lds(&Ks[cur][(nt * 16) * 64 + dk], 64);
        s[nt] = wmma16x16x32(aq, bk, s[nt]);
      }
    }

    // row max: in-register over 4 tiles, then packed-f16 shuffle reduction
    float rmv[8];
#pragma unroll
    for (int i = 0; i < 8; ++i)
      rmv[i] = fmaxf(fmaxf(s[0][i], s[1][i]), fmaxf(s[2][i], s[3][i]));
    unsigned pk[4];
#pragma unroll
    for (int p = 0; p < 4; ++p) pk[p] = pk_rtz(rmv[2 * p], rmv[2 * p + 1]);
#pragma unroll
    for (int m = 1; m < 16; m <<= 1) {
#pragma unroll
      for (int p = 0; p < 4; ++p)
        pk[p] = pk_max(pk[p], __shfl_xor(pk[p], m, 32));
    }

    // exp2-domain probabilities + packed b64 stores of P
#pragma unroll
    for (int i = 0; i < 8; ++i) {
      v2h hm = __builtin_bit_cast(v2h, pk[i >> 1]);
      float rm = (float)hm[i & 1];
      float mn = fmaxf(mS[i], rm);
      float al = exp2f(mS[i] - mn);
      float p0 = exp2f(s[0][i] - mn);
      float p1 = exp2f(s[1][i] - mn);
      float p2 = exp2f(s[2][i] - mn);
      float p3 = exp2f(s[3][i] - mn);
      alv[i] = al;
      mS[i]  = mn;
      acc[0][i] *= al; acc[1][i] *= al; acc[2][i] *= al; acc[3][i] *= al;
      uint2 u;
      u.x = pk_rtz(p0, p1);
      u.y = pk_rtz(p2, p3);
      *(uint2*)&Ps[wid][(halfb + i) * 64 + colo * 4] = u;   // keys 4c..4c+3
    }
    __syncthreads();

    // O += P @ V ; l += P @ ones  (row sums via WMMA, no shuffles)
    v16h ap0 = frag_a_lds(&Ps[wid][0], 64);
    v16h ap1 = frag_a_lds(&Ps[wid][32], 64);
    v8f lt = {};
    lt = wmma16x16x32(ap0, ones, lt);
    lt = wmma16x16x32(ap1, ones, lt);
#pragma unroll
    for (int t = 0; t < 4; ++t) {
      v16h bv0 = frag_bt_lds(&Vs[cur][(t * 16) * 64 + 0], 64);
      v16h bv1 = frag_bt_lds(&Vs[cur][(t * 16) * 64 + 32], 64);
      acc[t] = wmma16x16x32(ap0, bv0, acc[t]);
      acc[t] = wmma16x16x32(ap1, bv1, acc[t]);
    }
#pragma unroll
    for (int i = 0; i < 8; ++i) lS[i] = lS[i] * alv[i] + lt[i];
    __syncthreads();                    // readers done before buffer reuse
  }

  int b = bh >> 4;
  int h = bh & 15;
#pragma unroll
  for (int i = 0; i < 8; ++i) {
    float inv = 1.0f / lS[i];
    int q = qt * 64 + wid * 16 + halfb + i;
    size_t rowoff = ((size_t)(b * SEQ + q)) * CDIM + h * HDIM;
#pragma unroll
    for (int t = 0; t < 4; ++t) {
      Of[rowoff + t * 16 + colo] = (_Float16)(acc[t][i] * inv);
    }
  }
}

// ---------------------------------------------------------------------------
// Host launcher
// ---------------------------------------------------------------------------
extern "C" void kernel_launch(void* const* d_in, const int* in_sizes, int n_in,
                              void* d_out, int out_size, void* d_ws,
                              size_t ws_size, hipStream_t stream) {
  const float* x       = (const float*)d_in[0];
  const float* W_qkv   = (const float*)d_in[1];
  const float* b_qkv   = (const float*)d_in[2];
  const float* q_gamma = (const float*)d_in[3];
  const float* q_beta  = (const float*)d_in[4];
  const float* k_gamma = (const float*)d_in[5];
  const float* k_beta  = (const float*)d_in[6];
  const float* W_proj  = (const float*)d_in[7];
  const float* b_proj  = (const float*)d_in[8];

  char* ws = (char*)d_ws;
  size_t off = 0;
  float* qkv = (float*)(ws + off);  off += (size_t)NTOK * 3 * CDIM * 4;
  _Float16* xh  = (_Float16*)(ws + off); off += (size_t)NTOK * CDIM * 2;
  _Float16* Wqt = (_Float16*)(ws + off); off += (size_t)3 * CDIM * CDIM * 2;
  _Float16* Wpt = (_Float16*)(ws + off); off += (size_t)CDIM * CDIM * 2;
  _Float16* Qh  = (_Float16*)(ws + off); off += (size_t)NTOK * CDIM * 2;
  _Float16* Kh  = (_Float16*)(ws + off); off += (size_t)NTOK * CDIM * 2;
  _Float16* Vt  = (_Float16*)(ws + off); off += (size_t)NTOK * CDIM * 2;
  _Float16* Of  = (_Float16*)(ws + off); off += (size_t)NTOK * CDIM * 2;

  cvt_f16_kernel<<<dim3(NTOK * CDIM / 1024), 256, 0, stream>>>(x, xh);
  transpose_cvt_kernel<<<dim3((CDIM / 32) * (3 * CDIM / 32)), 256, 0, stream>>>(
      W_qkv, Wqt, CDIM, 3 * CDIM);
  transpose_cvt_kernel<<<dim3((CDIM / 32) * (CDIM / 32)), 256, 0, stream>>>(
      W_proj, Wpt, CDIM, CDIM);

  // 1) qkv = x @ W_qkv + b_qkv   (block tile 128x256)
  gemm_f16_bias_kernel<<<dim3((NTOK / 128) * (3 * CDIM / 256)), 256, 0,
                         stream>>>(xh, Wqt, b_qkv, qkv, NTOK, CDIM, 3 * CDIM);

  // 2) qk layernorm + scale + head split (V pre-transposed)
  qk_norm_kernel<<<dim3(NTOK), 256, 0, stream>>>(qkv, q_gamma, q_beta, k_gamma,
                                                 k_beta, Qh, Kh, Vt);

  // 3) fused flash attention -> Of (f16 [token][C])
  attn_kernel<<<dim3(NBATCH * NHEAD * (SEQ / 64)), 128, 0, stream>>>(Qh, Kh, Vt,
                                                                     Of);

  // 4) out = Of @ W_proj + b_proj
  gemm_f16_bias_kernel<<<dim3((NTOK / 128) * (CDIM / 256)), 256, 0, stream>>>(
      Of, Wpt, b_proj, (float*)d_out, NTOK, CDIM, CDIM);
}
